// Model_6279242187189
// MI455X (gfx1250) — matrix operand order repaired and checked by reading, not compiled, not run on hardware
//
#include <hip/hip_runtime.h>
#include <math.h>

// ---------------------------------------------------------------------------
// MI455X fused per-species MLP energy head (v3).
//
// Roofline: 41 GFLOP vs 252 MB feature reads -> ~11us at 23.3 TB/s HBM.
// BF16 WMMA (v_wmma_f32_16x16x32_bf16, f32 accum) keeps this memory-bound.
// 64-atom M-tiles (4 sub-tiles of 16 per wave) so every B fragment feeds
// 4 WMMAs (L2 weight traffic ~640 MB). v3 hoists the 4 A-fragment LDS loads
// per K-step into distinct registers so the 8 WMMAs issue as one burst
// instead of ds_load/s_wait_dscnt-serialized pairs.
// LDS: 61.4KB feat chunk + 2x32KB hidden acts ~= 127KB (< 320KB/WGP).
// ---------------------------------------------------------------------------

typedef __attribute__((ext_vector_type(16))) __bf16 v16bf;
typedef __attribute__((ext_vector_type(8)))  __bf16 v8bf;
typedef __attribute__((ext_vector_type(8)))  float  v8f;
typedef unsigned short u16;
typedef unsigned int   u32;

#define N_SPECIES 4
#define NFEAT     1920
#define HID       256
#define TILE_M    64             // atoms per block
#define MSUB      4              // 16-row WMMA sub-tiles per block
#define CHUNK     480            // K-chunk of features staged in LDS
#define SILU_GAIN 1.6765f
#define RS1       0.02282177322938192f   // 1/sqrt(1920)
#define RS2       0.0625f                // 1/sqrt(256)
#define ESCALE    0.0078125f             // (1/16) * 64^-0.5 = 1/128

static __device__ __forceinline__ u16 f2bf(float f) {
    u32 u = __float_as_uint(f);
    u32 r = u + 0x7FFFu + ((u >> 16) & 1u);   // round-to-nearest-even
    return (u16)(r >> 16);
}
static __device__ __forceinline__ float bf2f(u16 h) {
    return __uint_as_float(((u32)h) << 16);
}

static __device__ __forceinline__ v8f wmma_bf(v16bf a, v16bf b, v8f c) {
    // D = A(16x32 bf16) * B(32x16 bf16) + C(16x16 f32)
    return __builtin_amdgcn_wmma_f32_16x16x32_bf16(
        false, a, false, b, (short)0, c, false, false);
}

// A fragment: lane(0-15) = row M, lane>=16 same rows; per lane 16 contiguous
// K values starting at kbase + 16*half (matches loadB).
static __device__ __forceinline__ v16bf loadA(const __bf16* buf, int lane,
                                              int kbase, int stride) {
    int m = lane & 15, half = lane >> 4;
    const __bf16* p = buf + m * stride + kbase + 16 * half;
    return *(const v16bf*)p;
}

// B fragment from N-major bf16 weights [N][K]: lane = column N, 16 contiguous
// K values starting at kbase + 16*half (one aligned 32B load).
static __device__ __forceinline__ v16bf loadB(const __bf16* w, int lane,
                                              int n0, int kbase, int K) {
    int n = n0 + (lane & 15), half = lane >> 4;
    return *(const v16bf*)(w + (size_t)n * K + kbase + 16 * half);
}

// C layout: element j of lane l -> (M = j + 8*(l>>4), N = n0 + (l&15)).
static __device__ __forceinline__ void storeAct(u16* dst, v8f acc, int n0,
                                                int lane, float rs) {
    int nn = n0 + (lane & 15), half = lane >> 4;
#pragma unroll
    for (int j = 0; j < 8; ++j) {
        int m = j + 8 * half;
        float x = acc[j] * rs;
        float y = x / (1.f + __expf(-x)) * SILU_GAIN;   // silu * gain
        dst[m * HID + nn] = f2bf(y);
    }
}

// ---------------------------------------------------------------------------
// Small setup kernels
// ---------------------------------------------------------------------------

__global__ void kZero(float* out, int out_n, int* counters) {
    int i = blockIdx.x * blockDim.x + threadIdx.x;
    if (i < out_n) out[i] = 0.f;
    if (i < 12) counters[i] = 0;
}

__global__ void kCount(const int* __restrict__ species, int n, int* counters) {
    int i = blockIdx.x * blockDim.x + threadIdx.x;
    if (i < n) atomicAdd(&counters[species[i] & 3], 1);
}

__global__ void kScan(int* counters) {
    int off = 0;
    for (int s = 0; s < N_SPECIES; ++s) {
        counters[4 + s] = off;      // segment start
        counters[8 + s] = off;      // scatter cursor
        off += counters[s];
    }
}

__global__ void kScatter(const int* __restrict__ species, int n,
                         int* counters, int* __restrict__ perm) {
    int i = blockIdx.x * blockDim.x + threadIdx.x;
    if (i < n) {
        int pos = atomicAdd(&counters[8 + (species[i] & 3)], 1);
        perm[pos] = i;
    }
}

// f32 [S][K][N] -> bf16 [S][N][K]
__global__ void kRepack(const float* __restrict__ src, u16* __restrict__ dst,
                        int K, int N) {
    size_t total = (size_t)N_SPECIES * K * N;
    for (size_t i = (size_t)blockIdx.x * blockDim.x + threadIdx.x; i < total;
         i += (size_t)gridDim.x * blockDim.x) {
        size_t s = i / ((size_t)K * N);
        size_t rem = i - s * (size_t)K * N;
        size_t k = rem / N, n = rem - k * N;
        dst[(s * N + n) * K + k] = f2bf(src[i]);
    }
}

__global__ void kComp(const int* __restrict__ species,
                      const int* __restrict__ sidx,
                      const float* __restrict__ coeff, float* out, int n) {
    int i = blockIdx.x * blockDim.x + threadIdx.x;
    if (i < n) atomicAdd(&out[sidx[i]], coeff[species[i] & 3]);
}

// ---------------------------------------------------------------------------
// Fused 4-layer per-species MLP over one 64-atom tile.
// Block = 256 threads = 8 wave32s; wave w owns output columns [32w, 32w+32)
// across 4 M-subtiles (B fragments reused 4x).
// ---------------------------------------------------------------------------
__global__ __launch_bounds__(256) void kMlp(
    const float* __restrict__ feats, const int* __restrict__ counters,
    const int* __restrict__ perm, const int* __restrict__ sidx,
    const u16* __restrict__ w1b, const u16* __restrict__ w2b,
    const u16* __restrict__ w3b, const u16* __restrict__ w4b,
    float* __restrict__ out) {
    __shared__ u16 sFeat[TILE_M * CHUNK];   // 61440 B  bf16 feature chunk
    __shared__ u16 sH1[TILE_M * HID];       // 32768 B  h1 / h3
    __shared__ u16 sH2[TILE_M * HID];       // 32768 B  h2
    __shared__ int sAtom[TILE_M];

    const int s = blockIdx.y;
    const int t = blockIdx.x;
    const int count = counters[s];
    if (t * TILE_M >= count) return;
    const int base  = counters[4 + s] + t * TILE_M;
    const int valid = count - t * TILE_M < TILE_M ? count - t * TILE_M : TILE_M;

    const int tid = threadIdx.x;
    if (tid < TILE_M) sAtom[tid] = (tid < valid) ? perm[base + tid] : -1;

    const int lane = tid & 31;
    const int wave = tid >> 5;
    const int n0 = wave * 32;

    const __bf16* w1s = (const __bf16*)(w1b + (size_t)s * HID * NFEAT);
    const __bf16* w2s = (const __bf16*)(w2b + (size_t)s * HID * HID);
    const __bf16* w3s = (const __bf16*)(w3b + (size_t)s * HID * HID);

    // ---- Layer 1: [64 x 1920] x [1920 x 256], K streamed in 4 LDS chunks
    v8f acc1[MSUB][2];
#pragma unroll
    for (int mt = 0; mt < MSUB; ++mt) { acc1[mt][0] = {}; acc1[mt][1] = {}; }

    for (int chunk = 0; chunk < NFEAT / CHUNK; ++chunk) {
        __syncthreads();
        // stage + convert features (float4 global loads, 8B LDS stores)
        for (int idx = tid; idx < TILE_M * (CHUNK / 4); idx += 256) {
            int r = idx / (CHUNK / 4), c4 = idx % (CHUNK / 4);
            float4 v = make_float4(0.f, 0.f, 0.f, 0.f);
            int atom = sAtom[r];
            if (atom >= 0)
                v = ((const float4*)(feats + (size_t)atom * NFEAT +
                                     chunk * CHUNK))[c4];
            u32 p0 = (u32)f2bf(v.x) | ((u32)f2bf(v.y) << 16);
            u32 p1 = (u32)f2bf(v.z) | ((u32)f2bf(v.w) << 16);
            *(uint2*)&sFeat[r * CHUNK + c4 * 4] = make_uint2(p0, p1);
        }
        __syncthreads();
#pragma unroll
        for (int ks = 0; ks < CHUNK / 32; ++ks) {
            // hoist all A fragments into distinct registers, then burst WMMAs
            v16bf af[MSUB];
#pragma unroll
            for (int mt = 0; mt < MSUB; ++mt)
                af[mt] = loadA((const __bf16*)(sFeat + mt * 16 * CHUNK),
                               lane, ks * 32, CHUNK);
            v16bf b0 = loadB(w1s, lane, n0,      chunk * CHUNK + ks * 32, NFEAT);
            v16bf b1 = loadB(w1s, lane, n0 + 16, chunk * CHUNK + ks * 32, NFEAT);
#pragma unroll
            for (int mt = 0; mt < MSUB; ++mt) {
                acc1[mt][0] = wmma_bf(af[mt], b0, acc1[mt][0]);
                acc1[mt][1] = wmma_bf(af[mt], b1, acc1[mt][1]);
            }
        }
    }
#pragma unroll
    for (int mt = 0; mt < MSUB; ++mt) {
        storeAct(sH1 + mt * 16 * HID, acc1[mt][0], n0, lane, RS1);
        storeAct(sH1 + mt * 16 * HID, acc1[mt][1], n0 + 16, lane, RS1);
    }
    __syncthreads();

    // ---- Layer 2: [64 x 256] x [256 x 256]
    v8f acc2[MSUB][2];
#pragma unroll
    for (int mt = 0; mt < MSUB; ++mt) { acc2[mt][0] = {}; acc2[mt][1] = {}; }
#pragma unroll
    for (int ks = 0; ks < HID / 32; ++ks) {
        v16bf af[MSUB];
#pragma unroll
        for (int mt = 0; mt < MSUB; ++mt)
            af[mt] = loadA((const __bf16*)(sH1 + mt * 16 * HID),
                           lane, ks * 32, HID);
        v16bf b0 = loadB(w2s, lane, n0,      ks * 32, HID);
        v16bf b1 = loadB(w2s, lane, n0 + 16, ks * 32, HID);
#pragma unroll
        for (int mt = 0; mt < MSUB; ++mt) {
            acc2[mt][0] = wmma_bf(af[mt], b0, acc2[mt][0]);
            acc2[mt][1] = wmma_bf(af[mt], b1, acc2[mt][1]);
        }
    }
#pragma unroll
    for (int mt = 0; mt < MSUB; ++mt) {
        storeAct(sH2 + mt * 16 * HID, acc2[mt][0], n0, lane, RS2);
        storeAct(sH2 + mt * 16 * HID, acc2[mt][1], n0 + 16, lane, RS2);
    }
    __syncthreads();

    // ---- Layer 3: [64 x 256] x [256 x 256]  (writes back into sH1)
    v8f acc3[MSUB][2];
#pragma unroll
    for (int mt = 0; mt < MSUB; ++mt) { acc3[mt][0] = {}; acc3[mt][1] = {}; }
#pragma unroll
    for (int ks = 0; ks < HID / 32; ++ks) {
        v16bf af[MSUB];
#pragma unroll
        for (int mt = 0; mt < MSUB; ++mt)
            af[mt] = loadA((const __bf16*)(sH2 + mt * 16 * HID),
                           lane, ks * 32, HID);
        v16bf b0 = loadB(w3s, lane, n0,      ks * 32, HID);
        v16bf b1 = loadB(w3s, lane, n0 + 16, ks * 32, HID);
#pragma unroll
        for (int mt = 0; mt < MSUB; ++mt) {
            acc3[mt][0] = wmma_bf(af[mt], b0, acc3[mt][0]);
            acc3[mt][1] = wmma_bf(af[mt], b1, acc3[mt][1]);
        }
    }
#pragma unroll
    for (int mt = 0; mt < MSUB; ++mt) {
        storeAct(sH1 + mt * 16 * HID, acc3[mt][0], n0, lane, RS2);
        storeAct(sH1 + mt * 16 * HID, acc3[mt][1], n0 + 16, lane, RS2);
    }
    __syncthreads();

    // ---- Layer 4: [64 x 256] x [256 x 1] dot + segment-sum scatter
    if (tid < TILE_M && sAtom[tid] >= 0) {
        const u16* w4s = w4b + s * HID;
        float acc = 0.f;
#pragma unroll 8
        for (int k = 0; k < HID; ++k)
            acc += bf2f(sH1[tid * HID + k]) * bf2f(w4s[k]);
        atomicAdd(&out[sidx[sAtom[tid]]], acc * ESCALE);
    }
}

// ---------------------------------------------------------------------------
// Launch
// ---------------------------------------------------------------------------
extern "C" void kernel_launch(void* const* d_in, const int* in_sizes, int n_in,
                              void* d_out, int out_size, void* d_ws,
                              size_t ws_size, hipStream_t stream) {
    const float* feats  = (const float*)d_in[0];
    const float* W1     = (const float*)d_in[1];
    const float* W2     = (const float*)d_in[2];
    const float* W3     = (const float*)d_in[3];
    const float* W4     = (const float*)d_in[4];
    const float* coeff  = (const float*)d_in[5];
    const int*   species= (const int*)d_in[6];
    const int*   sidx   = (const int*)d_in[7];
    const int n = in_sizes[6];                 // 32768 atoms
    float* out = (float*)d_out;

    // workspace carve-up (base is 256B aligned from hipMalloc)
    char* w = (char*)d_ws;
    int* counters = (int*)w;                   // [0:4) counts, [4:8) offs, [8:12) cursors
    int* perm     = (int*)(w + 64);
    u16* w1b = (u16*)(w + 64 + (size_t)n * 4); // bf16 [S][N][K], 32B-aligned
    u16* w2b = w1b + (size_t)N_SPECIES * HID * NFEAT;
    u16* w3b = w2b + (size_t)N_SPECIES * HID * HID;
    u16* w4b = w3b + (size_t)N_SPECIES * HID * HID;

    const int blk = 256;
    const int gAtoms = (n + blk - 1) / blk;

    kZero<<<(out_size + blk - 1) / blk + 1, blk, 0, stream>>>(out, out_size, counters);
    kCount<<<gAtoms, blk, 0, stream>>>(species, n, counters);
    kScan<<<1, 1, 0, stream>>>(counters);
    kScatter<<<gAtoms, blk, 0, stream>>>(species, n, counters, perm);

    kRepack<<<(N_SPECIES * NFEAT * HID + blk - 1) / blk, blk, 0, stream>>>(W1, w1b, NFEAT, HID);
    kRepack<<<(N_SPECIES * HID * HID + blk - 1) / blk, blk, 0, stream>>>(W2, w2b, HID, HID);
    kRepack<<<(N_SPECIES * HID * HID + blk - 1) / blk, blk, 0, stream>>>(W3, w3b, HID, HID);
    kRepack<<<(N_SPECIES * HID + blk - 1) / blk, blk, 0, stream>>>(W4, w4b, HID, 1);

    dim3 grid((n + TILE_M - 1) / TILE_M, N_SPECIES);
    kMlp<<<grid, blk, 0, stream>>>(feats, counters, perm, sidx,
                                   w1b, w2b, w3b, w4b, out);

    kComp<<<gAtoms, blk, 0, stream>>>(species, sidx, coeff, out, n);
}
